// CGCNN_34866544509578
// MI455X (gfx1250) — matrix operand order repaired and checked by reading, not compiled, not run on hardware
//
#include <hip/hip_runtime.h>
#include <hip/hip_bf16.h>
#include <math.h>

// ---------------- problem constants (match reference) ----------------
#define N_NODES  50000
#define N_EDGES  400000
#define N_GRAPHS 512
#define NF       64      // node features
#define EF       32      // edge features
#define AIF      92      // atom input features
#define FCF      128     // fc features
#define NL       3       // conv layers
#define BN_EPS   1e-5f

typedef __attribute__((ext_vector_type(16))) __bf16 v16bf;
typedef __attribute__((ext_vector_type(8)))  float  v8f;

__device__ __forceinline__ float softplus_f(float x) {
    return x > 20.0f ? x : log1pf(__expf(x));
}
__device__ __forceinline__ float sigmoid_f(float x) {
    return 1.0f / (1.0f + __expf(-x));
}

// ---------------- small utility kernels ----------------
__global__ void k_zero(float* __restrict__ p, int n) {
    int i = blockIdx.x * blockDim.x + threadIdx.x;
    if (i < n) p[i] = 0.0f;
}

// RBF expansion -> bf16 edge features [E, EF]
__global__ void k_rbf(const float* __restrict__ d, __bf16* __restrict__ ebf, int total) {
    int i = blockIdx.x * blockDim.x + threadIdx.x;
    if (i >= total) return;
    int e = i >> 5, j = i & 31;
    float c = (float)j * (8.0f / 31.0f);          // linspace(0,8,32)
    float x = d[e] - c;
    ebf[i] = (__bf16)__expf(-(31.0f / 8.0f) * x * x);  // gamma = 1/(8/31)
}

// h = atom_features @ W_emb + b_emb ; also emit bf16 copy for WMMA gathers
__global__ void k_embed(const float* __restrict__ af, const float* __restrict__ W,
                        const float* __restrict__ b, float* __restrict__ h,
                        __bf16* __restrict__ hbf, int total) {
    int i = blockIdx.x * blockDim.x + threadIdx.x;
    if (i >= total) return;
    int n = i >> 6, f = i & 63;
    const float* row = af + (size_t)n * AIF;
    float s = b[f];
    #pragma unroll 4
    for (int k = 0; k < AIF; ++k) s += row[k] * W[k * NF + f];
    h[i]   = s;
    hbf[i] = (__bf16)s;
}

// Swizzle Wi[l]/Wu[l] ([160,64] f32) into bf16 B-operand fragments:
// Bswz[((mat*5 + s)*4 + t)*32 + lane][j]  (lane-major, 16 bf16 = 32B per lane)
// lane<16 : N = t*16+lane, K = s*32 + (j&7) + (j>=8 ? 16 : 0)      (j = 0..15)
// lane>=16: N = t*16+(lane-16), K = s*32+8+(j&7) + (j>=8 ? 16 : 0)
__global__ void k_cvt_w(const float* __restrict__ Wi, const float* __restrict__ Wu,
                        __bf16* __restrict__ Bswz) {
    int i = blockIdx.x * blockDim.x + threadIdx.x;
    if (i >= 2 * 5 * 4 * 32 * 16) return;
    int j    = i & 15;
    int lane = (i >> 4) & 31;
    int t    = (i >> 9) & 3;
    int rest = i >> 11;
    int s    = rest % 5;
    int mat  = rest / 5;
    int n = t * 16 + (lane & 15);
    int k = s * 32 + ((lane >> 4) ? 8 : 0) + (j & 7) + ((j & 8) ? 16 : 0);
    const float* W = mat ? Wu : Wi;
    Bswz[i] = (__bf16)W[k * NF + n];
}

// ---------------- WMMA edge GEMM (+ fused BN column statistics) ----------------
// One wave per 16-edge tile: yi = z @ Wi, yu = z @ Wu (bias dropped: BN cancels it).
// z-row (bf16): [ h[src] (64) | h[dst] (64) | e (32) ] = 320 bytes, staged in LDS.
// Epilogue computes per-column sum / sum-of-squares from the live accumulators and
// atomically folds one partial per block into the global stats buffer.
union Frag32B { uint4 u[2]; v16bf v; };

__global__ __launch_bounds__(256) void k_edge_gemm(
    const __bf16* __restrict__ hbf, const __bf16* __restrict__ ebf,
    const int* __restrict__ src, const int* __restrict__ dst,
    const __bf16* __restrict__ Bswz,
    float* __restrict__ yi, float* __restrict__ yu,
    float* __restrict__ sums /* [0..63]=sumI [64..127]=sqI [128..191]=sumU [192..255]=sqU */)
{
    __shared__ __align__(16) unsigned char smem[8 * 16 * 320];  // 8 waves x 5KB
    const int wave = threadIdx.x >> 5;
    const int lane = threadIdx.x & 31;
    const int tile = blockIdx.x * 8 + wave;
    unsigned char* zbase = smem + wave * (16 * 320);

    // Stage: 2 lanes per edge copy the 320B z-row with b128 transfers.
    {
        int eidx = lane >> 1, half = lane & 1;
        int e = tile * 16 + eidx;
        const uint4* hs = (const uint4*)(hbf + (size_t)src[e] * NF);
        const uint4* hd = (const uint4*)(hbf + (size_t)dst[e] * NF);
        const uint4* ee = (const uint4*)(ebf + (size_t)e * EF);
        uint4* zrow = (uint4*)(zbase + eidx * 320);   // 20 x 16B chunks
        if (half == 0) {
            #pragma unroll
            for (int c = 0; c < 8; ++c) zrow[c] = hs[c];
            zrow[8] = hd[0]; zrow[9] = hd[1];
        } else {
            #pragma unroll
            for (int c = 0; c < 6; ++c) zrow[10 + c] = hd[2 + c];
            #pragma unroll
            for (int c = 0; c < 4; ++c) zrow[16 + c] = ee[c];
        }
    }
    __syncthreads();

    v8f accI[4], accU[4];
    #pragma unroll
    for (int t = 0; t < 4; ++t)
        #pragma unroll
        for (int q = 0; q < 8; ++q) { accI[t][q] = 0.0f; accU[t][q] = 0.0f; }

    const int m  = lane & 15;
    const int hi = lane >> 4;

    #pragma unroll
    for (int s = 0; s < 5; ++s) {
        // A fragment (16-bit A 16x32 layout): lanes lo K=[s*32+0..7, +16..23],
        // lanes hi K=[s*32+8..15, +24..31] -> two 16B LDS loads.
        Frag32B a;
        const unsigned char* ab = zbase + m * 320 + s * 64 + hi * 16;
        a.u[0] = *(const uint4*)ab;
        a.u[1] = *(const uint4*)(ab + 32);
        #pragma unroll
        for (int t = 0; t < 4; ++t) {
            Frag32B bI, bU;
            const uint4* bpI = (const uint4*)(Bswz + (size_t)((((0 * 5 + s) * 4 + t) * 32 + lane) * 16));
            const uint4* bpU = (const uint4*)(Bswz + (size_t)((((1 * 5 + s) * 4 + t) * 32 + lane) * 16));
            bI.u[0] = bpI[0]; bI.u[1] = bpI[1];
            bU.u[0] = bpU[0]; bU.u[1] = bpU[1];
            accI[t] = __builtin_amdgcn_wmma_f32_16x16x32_bf16(
                false, a.v, false, bI.v, (short)0, accI[t], false, false);
            accU[t] = __builtin_amdgcn_wmma_f32_16x16x32_bf16(
                false, a.v, false, bU.v, (short)0, accU[t], false, false);
        }
    }

    // C/D layout: VGPR r -> M = r + hi*8, N = t*16 + (lane&15). 64B-contiguous stores.
    #pragma unroll
    for (int t = 0; t < 4; ++t)
        #pragma unroll
        for (int r = 0; r < 8; ++r) {
            size_t row = (size_t)tile * 16 + r + hi * 8;
            yi[row * NF + t * 16 + m] = accI[t][r];
            yu[row * NF + t * 16 + m] = accU[t][r];
        }

    // ---- fused BN statistics (reuse staging LDS after barrier) ----
    __syncthreads();
    float* red = (float*)smem;   // [4][8 waves][64 cols] = 8KB
    #pragma unroll
    for (int t = 0; t < 4; ++t) {
        float sI = 0.f, qI = 0.f, sU = 0.f, qU = 0.f;
        #pragma unroll
        for (int r = 0; r < 8; ++r) {
            float vi = accI[t][r], vu = accU[t][r];
            sI += vi; qI += vi * vi;
            sU += vu; qU += vu * vu;
        }
        // column f = t*16+m lives in lanes m (rows 0-7) and m+16 (rows 8-15)
        sI += __shfl_xor(sI, 16, 32);
        qI += __shfl_xor(qI, 16, 32);
        sU += __shfl_xor(sU, 16, 32);
        qU += __shfl_xor(qU, 16, 32);
        if (hi == 0) {
            int f = t * 16 + m;
            red[0 * 512 + wave * 64 + f] = sI;
            red[1 * 512 + wave * 64 + f] = qI;
            red[2 * 512 + wave * 64 + f] = sU;
            red[3 * 512 + wave * 64 + f] = qU;
        }
    }
    __syncthreads();
    if (threadIdx.x < 64) {
        int f = threadIdx.x;
        float sI = 0.f, qI = 0.f, sU = 0.f, qU = 0.f;
        #pragma unroll
        for (int w = 0; w < 8; ++w) {
            sI += red[0 * 512 + w * 64 + f];
            qI += red[1 * 512 + w * 64 + f];
            sU += red[2 * 512 + w * 64 + f];
            qU += red[3 * 512 + w * 64 + f];
        }
        atomicAdd(&sums[f],        sI);
        atomicAdd(&sums[64  + f],  qI);
        atomicAdd(&sums[128 + f],  sU);
        atomicAdd(&sums[192 + f],  qU);
    }
}

// ---------------- BatchNorm statistics (used for agg only) ----------------
__global__ __launch_bounds__(256) void k_colstats(const float* __restrict__ y, int rows,
                                                  float* __restrict__ sums) {
    int f = threadIdx.x & 63;
    int sub = threadIdx.x >> 6;                    // 0..3
    float s = 0.0f, q = 0.0f;
    for (int r = blockIdx.x * 4 + sub; r < rows; r += gridDim.x * 4) {
        float v = y[(size_t)r * NF + f];
        s += v; q += v * v;
    }
    __shared__ float ls[256], lq[256];
    ls[threadIdx.x] = s; lq[threadIdx.x] = q;
    __syncthreads();
    if (sub == 0) {
        s = ls[f] + ls[64 + f] + ls[128 + f] + ls[192 + f];
        q = lq[f] + lq[64 + f] + lq[128 + f] + lq[192 + f];
        atomicAdd(&sums[f], s);
        atomicAdd(&sums[64 + f], q);
    }
}

// Fold BN into scale/shift:  bn(x) = x*scale + shift
__global__ void k_bn_fin(const float* __restrict__ sums, float cnt,
                         const float* __restrict__ gamma, const float* __restrict__ beta,
                         float* __restrict__ scale, float* __restrict__ shift) {
    int f = threadIdx.x;
    if (f >= NF) return;
    float mean = sums[f] / cnt;
    float var  = sums[64 + f] / cnt - mean * mean;   // biased, matches torch
    float sc   = gamma[f] * rsqrtf(var + BN_EPS);
    scale[f] = sc;
    shift[f] = beta[f] - mean * sc;
}

// gate*upd per edge element (x4 vectorized), scatter-add into agg[dst]
__global__ void k_msg(const float4* __restrict__ yi4, const float4* __restrict__ yu4,
                      const float* __restrict__ scI, const float* __restrict__ shI,
                      const float* __restrict__ scU, const float* __restrict__ shU,
                      const int* __restrict__ dst, float* __restrict__ agg, int total) {
    int i = blockIdx.x * blockDim.x + threadIdx.x;   // total = E * 16
    if (i >= total) return;
    int e  = i >> 4;
    int f0 = (i & 15) * 4;
    float4 vi = yi4[i];
    float4 vu = yu4[i];
    float* ap = &agg[(size_t)dst[e] * NF + f0];
    atomicAdd(&ap[0], sigmoid_f(vi.x * scI[f0+0] + shI[f0+0]) * softplus_f(vu.x * scU[f0+0] + shU[f0+0]));
    atomicAdd(&ap[1], sigmoid_f(vi.y * scI[f0+1] + shI[f0+1]) * softplus_f(vu.y * scU[f0+1] + shU[f0+1]));
    atomicAdd(&ap[2], sigmoid_f(vi.z * scI[f0+2] + shI[f0+2]) * softplus_f(vu.z * scU[f0+2] + shU[f0+2]));
    atomicAdd(&ap[3], sigmoid_f(vi.w * scI[f0+3] + shI[f0+3]) * softplus_f(vu.w * scU[f0+3] + shU[f0+3]));
}

// h = softplus(h + bn(agg)); refresh bf16 copy
__global__ void k_node_upd(float* __restrict__ h, __bf16* __restrict__ hbf,
                           const float* __restrict__ agg,
                           const float* __restrict__ sc, const float* __restrict__ sh,
                           int total) {
    int i = blockIdx.x * blockDim.x + threadIdx.x;
    if (i >= total) return;
    int f = i & 63;
    float v = softplus_f(h[i] + agg[i] * sc[f] + sh[f]);
    h[i]   = v;
    hbf[i] = (__bf16)v;
}

// ---------------- pooling + head ----------------
__global__ void k_pool(const float* __restrict__ h, const int* __restrict__ gid,
                       float* __restrict__ pooled, float* __restrict__ cnts, int total) {
    int i = blockIdx.x * blockDim.x + threadIdx.x;
    if (i >= total) return;
    int n = i >> 6, f = i & 63;
    int g = gid[n];
    atomicAdd(&pooled[(size_t)g * NF + f], h[i]);
    if (f == 0) atomicAdd(&cnts[g], 1.0f);
}

__global__ void k_pool_fin(float* __restrict__ pooled, const float* __restrict__ cnts, int total) {
    int i = blockIdx.x * blockDim.x + threadIdx.x;
    if (i >= total) return;
    int g = i >> 6;
    float c = fmaxf(cnts[g], 1.0f);
    pooled[i] = softplus_f(pooled[i] / c);   // f = softplus(pooled/count)
}

__global__ void k_fc(const float* __restrict__ pooled, const float* __restrict__ Wfc,
                     const float* __restrict__ bfc, float* __restrict__ fbuf, int total) {
    int i = blockIdx.x * blockDim.x + threadIdx.x;
    if (i >= total) return;
    int g = i >> 7, c = i & 127;
    float acc = bfc[c];
    #pragma unroll 4
    for (int k = 0; k < NF; ++k) acc += pooled[(size_t)g * NF + k] * Wfc[k * FCF + c];
    fbuf[i] = softplus_f(softplus_f(acc));   // softplus(fc) then extra softplus
}

__global__ void k_out(const float* __restrict__ fbuf, const float* __restrict__ Wout,
                      const float* __restrict__ bout, float* __restrict__ out, int total) {
    int g = blockIdx.x * blockDim.x + threadIdx.x;
    if (g >= total) return;
    float s = bout[0];
    #pragma unroll 4
    for (int c = 0; c < FCF; ++c) s += fbuf[(size_t)g * FCF + c] * Wout[c];
    out[g] = s;
}

// ---------------- launcher ----------------
extern "C" void kernel_launch(void* const* d_in, const int* in_sizes, int n_in,
                              void* d_out, int out_size, void* d_ws, size_t ws_size,
                              hipStream_t stream) {
    (void)in_sizes; (void)n_in; (void)out_size; (void)ws_size;
    const float* af    = (const float*)d_in[0];
    const float* bond  = (const float*)d_in[1];
    const int*   src   = (const int*)  d_in[2];
    const int*   dst   = (const int*)  d_in[3];
    const int*   gid   = (const int*)  d_in[4];
    const float* W_emb = (const float*)d_in[5];
    const float* b_emb = (const float*)d_in[6];
    const float* Wi    = (const float*)d_in[7];
    // bi (d_in[8]) and bu (d_in[12]) cancel exactly in training-mode BatchNorm.
    const float* gi    = (const float*)d_in[9];
    const float* bti   = (const float*)d_in[10];
    const float* Wu    = (const float*)d_in[11];
    const float* gu    = (const float*)d_in[13];
    const float* btu   = (const float*)d_in[14];
    const float* g_bn  = (const float*)d_in[15];
    const float* b_bn  = (const float*)d_in[16];
    const float* W_fc  = (const float*)d_in[17];
    const float* b_fc  = (const float*)d_in[18];
    const float* W_out = (const float*)d_in[19];
    const float* b_out = (const float*)d_in[20];
    float* out = (float*)d_out;

    char* ws = (char*)d_ws;
    size_t off = 0;
    auto take = [&](size_t bytes) -> char* {
        char* p = ws + off;
        off = (off + bytes + 255) & ~(size_t)255;
        return p;
    };
    __bf16* ebf    = (__bf16*)take((size_t)N_EDGES * EF * 2);
    float*  h      = (float*) take((size_t)N_NODES * NF * 4);
    __bf16* hbf    = (__bf16*)take((size_t)N_NODES * NF * 2);
    float*  yi     = (float*) take((size_t)N_EDGES * NF * 4);
    float*  yu     = (float*) take((size_t)N_EDGES * NF * 4);
    float*  agg    = (float*) take((size_t)N_NODES * NF * 4);
    __bf16* Bswz   = (__bf16*)take((size_t)2 * 5 * 4 * 32 * 16 * 2);
    float*  st     = (float*) take(768 * 4);
    float*  pooled = (float*) take((size_t)N_GRAPHS * NF * 4);
    float*  cnts   = (float*) take((size_t)N_GRAPHS * 4);
    float*  fbuf   = (float*) take((size_t)N_GRAPHS * FCF * 4);

    // stats region layout (floats):
    // [0..63] sumI [64..127] sqI [128..191] sumU [192..255] sqU [256..383] sumsA
    // [384] scI [448] shI [512] scU [576] shU [640] scA [704] shA

    k_rbf<<<(N_EDGES * EF + 255) / 256, 256, 0, stream>>>(bond, ebf, N_EDGES * EF);
    k_embed<<<(N_NODES * NF + 255) / 256, 256, 0, stream>>>(af, W_emb, b_emb, h, hbf,
                                                            N_NODES * NF);

    for (int l = 0; l < NL; ++l) {
        k_cvt_w<<<80, 256, 0, stream>>>(Wi + (size_t)l * 160 * NF,
                                        Wu + (size_t)l * 160 * NF, Bswz);
        k_zero<<<(N_NODES * NF + 255) / 256, 256, 0, stream>>>(agg, N_NODES * NF);
        k_zero<<<2, 256, 0, stream>>>(st, 384);

        // GEMM + fused BN stats for yi/yu
        k_edge_gemm<<<N_EDGES / 16 / 8, 256, 0, stream>>>(hbf, ebf, src, dst, Bswz,
                                                          yi, yu, st);

        k_bn_fin<<<1, 64, 0, stream>>>(st,       (float)N_EDGES, gi  + l * NF, bti + l * NF,
                                       st + 384, st + 448);
        k_bn_fin<<<1, 64, 0, stream>>>(st + 128, (float)N_EDGES, gu  + l * NF, btu + l * NF,
                                       st + 512, st + 576);

        k_msg<<<(N_EDGES * 16 + 255) / 256, 256, 0, stream>>>(
            (const float4*)yi, (const float4*)yu, st + 384, st + 448,
            st + 512, st + 576, dst, agg, N_EDGES * 16);

        k_colstats<<<512, 256, 0, stream>>>(agg, N_NODES, st + 256);
        k_bn_fin<<<1, 64, 0, stream>>>(st + 256, (float)N_NODES, g_bn + l * NF, b_bn + l * NF,
                                       st + 640, st + 704);

        k_node_upd<<<(N_NODES * NF + 255) / 256, 256, 0, stream>>>(h, hbf, agg, st + 640,
                                                                   st + 704, N_NODES * NF);
    }

    k_zero<<<(N_GRAPHS * NF + 255) / 256, 256, 0, stream>>>(pooled, N_GRAPHS * NF);
    k_zero<<<1, 256, 0, stream>>>(cnts, N_GRAPHS);
    k_pool<<<(N_NODES * NF + 255) / 256, 256, 0, stream>>>(h, gid, pooled, cnts, N_NODES * NF);
    k_pool_fin<<<(N_GRAPHS * NF + 255) / 256, 256, 0, stream>>>(pooled, cnts, N_GRAPHS * NF);
    k_fc<<<(N_GRAPHS * FCF + 255) / 256, 256, 0, stream>>>(pooled, W_fc, b_fc, fbuf,
                                                           N_GRAPHS * FCF);
    k_out<<<(N_GRAPHS + 255) / 256, 256, 0, stream>>>(fbuf, W_out, b_out, out, N_GRAPHS);
}
